// BiLSTM_CRF_44358422233229
// MI455X (gfx1250) — matrix operand order repaired and checked by reading, not compiled, not run on hardware
//
#include <hip/hip_runtime.h>

typedef __attribute__((ext_vector_type(16))) _Float16 v16h;
typedef __attribute__((ext_vector_type(8)))  _Float16 v8h;
typedef __attribute__((ext_vector_type(8)))  float    v8f;

#define K_TAG     48
#define START_TAG 46
#define STOP_TAG  47
#define ROWSTRIDE 72   // f16 units: 144 bytes/row -> 16B-aligned rows, no bank conflicts

#define LOG2E   1.4426950408889634f
#define INVLN2  1.4426950408889634f
#define LN2     0.6931471805599453f

static __device__ __forceinline__ void lds_sync_wave() {
    // single-wave block: just order LDS ops, no s_barrier needed
    __builtin_amdgcn_wave_barrier();
    asm volatile("s_wait_dscnt 0" ::: "memory");
    __builtin_amdgcn_wave_barrier();
}

// CT = compile-time T (0 -> runtime). Base-2 log domain throughout:
//   g[b][i]     = fv[b][i] / ln2
//   p[b][j]     = 2^(g[b][j] - m[b])                       (f16, in LDS)
//   E[i][j]     = 2^(trans[i][j] * log2e)                  (f16, in regs)
//   g_new[b][i] = fma(feat, 1/ln2, m[b] + log2(sum_j p*E))
template<int CT>
__global__ __launch_bounds__(32)
void crf_fwd_wmma_kernel(const float* __restrict__ feats,
                         const float* __restrict__ trans,
                         float* __restrict__ out, int Trt)
{
    const int T = CT ? CT : Trt;
    __shared__ __align__(16) _Float16 pbuf[16 * ROWSTRIDE];

    const int lane = threadIdx.x & 31;
    const int n    = lane & 15;   // tag-within-tile / column lane
    const int half = lane >> 4;   // 0: batches r, 1: batches r+8 (D layout)
    const int b0   = blockIdx.x * 16;

    // ---- init p buffer: p[b][j] = (j==START) ? 1 : 0  (g = -inf-ish except START=0, m=0)
    {
        unsigned int* pz = (unsigned int*)pbuf;
        #pragma unroll
        for (int k = lane; k < (16 * ROWSTRIDE) / 2; k += 32) pz[k] = 0u;
        lds_sync_wave();
        if (half == 0) pbuf[n * ROWSTRIDE + START_TAG] = (_Float16)1.0f;
        lds_sync_wave();
    }

    // ---- B fragments: E^T = exp(transitions), ISA 16-bit B layout, K padded to 64
    // element e of fragment (tile, h) holds j = 32*h + 8*half + (e&7) + 16*(e>>3) for column i = 16*tile + n
    v16h B[3][2];
    #pragma unroll
    for (int tile = 0; tile < 3; ++tile) {
        const int i = tile * 16 + n;
        #pragma unroll
        for (int h = 0; h < 2; ++h) {
            v16h f;
            #pragma unroll
            for (int e = 0; e < 16; ++e) {
                const int j = 32 * h + 8 * half + (e & 7) + 16 * (e >> 3);
                const float v = (j < K_TAG)
                    ? __builtin_amdgcn_exp2f(trans[i * K_TAG + j] * LOG2E) : 0.0f;
                f[e] = (_Float16)v;
            }
            B[tile][h] = f;
        }
    }

    float m[8];
    #pragma unroll
    for (int r = 0; r < 8; ++r) m[r] = 0.0f;
    float g[8][3];

    // feats[(b)*T*48 + t*48 + i] ; b = b0 + 8*half + r ; i = 16*tile + n
    const float* fp = feats + ((size_t)(b0 + 8 * half) * (size_t)T) * K_TAG + (size_t)n;
    const size_t rstr = (size_t)T * K_TAG;   // compile-time when CT != 0

    // software-pipelined feat buffers (one step of global-latency hiding)
    float ftc[8][3], ftn[8][3];
    #pragma unroll
    for (int r = 0; r < 8; ++r)
        #pragma unroll
        for (int tile = 0; tile < 3; ++tile)
            ftc[r][tile] = fp[(size_t)r * rstr + tile * 16];

    for (int t = 0; t < T; ++t) {
        // prefetch feats for t+1 (uniform clamp; immediate offsets when CT!=0)
        const size_t tn = (size_t)((t + 1 < T) ? t + 1 : t) * K_TAG;
        #pragma unroll
        for (int r = 0; r < 8; ++r)
            #pragma unroll
            for (int tile = 0; tile < 3; ++tile)
                ftn[r][tile] = fp[(size_t)r * rstr + tn + tile * 16];

        // A fragments (16-bit A layout): lane row = n, K sub-blocks shifted by 8*half
        const _Float16* row = pbuf + n * ROWSTRIDE;
        v8h a0lo = *(const v8h*)(row +       8 * half);   // VGPR0-3 of frag0
        v8h a0hi = *(const v8h*)(row + 16  + 8 * half);   // VGPR4-7 of frag0
        v8h a1lo = *(const v8h*)(row + 32  + 8 * half);   // VGPR0-3 of frag1
        v8h a1hi = *(const v8h*)(row + 48  + 8 * half);   // VGPR4-7 of frag1
        v16h A0 = __builtin_shufflevector(a0lo, a0hi, 0,1,2,3,4,5,6,7,8,9,10,11,12,13,14,15);
        v16h A1 = __builtin_shufflevector(a1lo, a1hi, 0,1,2,3,4,5,6,7,8,9,10,11,12,13,14,15);

        // s[b][i] = sum_j p[b][j] * E[i][j] : three 16x16 output tiles, K = 64
        v8f c[3];
        #pragma unroll
        for (int tile = 0; tile < 3; ++tile) {
            v8f acc = {};
            acc = __builtin_amdgcn_wmma_f32_16x16x32_f16(false, A0, false, B[tile][0],
                                                         (short)0, acc, false, false);
            acc = __builtin_amdgcn_wmma_f32_16x16x32_f16(false, A1, false, B[tile][1],
                                                         (short)0, acc, false, false);
            c[tile] = acc;
        }

        // g = log2(s) + m + feat/ln2 ; new running max via xor-shuffles within 16-lane half
        float nm[8];
        #pragma unroll
        for (int r = 0; r < 8; ++r) {
            const float v0 = fmaf(ftc[r][0], INVLN2, __builtin_amdgcn_logf(c[0][r]) + m[r]);
            const float v1 = fmaf(ftc[r][1], INVLN2, __builtin_amdgcn_logf(c[1][r]) + m[r]);
            const float v2 = fmaf(ftc[r][2], INVLN2, __builtin_amdgcn_logf(c[2][r]) + m[r]);
            g[r][0] = v0; g[r][1] = v1; g[r][2] = v2;
            float mx = fmaxf(v0, fmaxf(v1, v2));
            #pragma unroll
            for (int s = 1; s < 16; s <<= 1)
                mx = fmaxf(mx, __shfl_xor(mx, s, 32));
            nm[r] = mx;
        }

        // p = 2^(g - m_new) -> f16 back into LDS for the next step's A
        #pragma unroll
        for (int r = 0; r < 8; ++r) {
            _Float16* prow = pbuf + (r + 8 * half) * ROWSTRIDE;
            #pragma unroll
            for (int tile = 0; tile < 3; ++tile)
                prow[tile * 16 + n] = (_Float16)__builtin_amdgcn_exp2f(g[r][tile] - nm[r]);
            m[r] = nm[r];
        }
        lds_sync_wave();

        // rotate feat buffers
        #pragma unroll
        for (int r = 0; r < 8; ++r)
            #pragma unroll
            for (int tile = 0; tile < 3; ++tile)
                ftc[r][tile] = ftn[r][tile];
    }

    // ---- terminal: logZ[b] = ln2 * log2sumexp2_i( g[b][i] + trans[STOP][i]/ln2 )
    float tstop[3];
    #pragma unroll
    for (int tile = 0; tile < 3; ++tile)
        tstop[tile] = trans[STOP_TAG * K_TAG + tile * 16 + n] * INVLN2;

    #pragma unroll
    for (int r = 0; r < 8; ++r) {
        const float v0 = g[r][0] + tstop[0];
        const float v1 = g[r][1] + tstop[1];
        const float v2 = g[r][2] + tstop[2];
        float mx = fmaxf(v0, fmaxf(v1, v2));
        #pragma unroll
        for (int s = 1; s < 16; s <<= 1)
            mx = fmaxf(mx, __shfl_xor(mx, s, 32));
        float sm = __builtin_amdgcn_exp2f(v0 - mx) + __builtin_amdgcn_exp2f(v1 - mx)
                 + __builtin_amdgcn_exp2f(v2 - mx);
        #pragma unroll
        for (int s = 1; s < 16; s <<= 1)
            sm += __shfl_xor(sm, s, 32);
        if (n == 0)
            out[b0 + r + 8 * half] = LN2 * (mx + __builtin_amdgcn_logf(sm));
    }
}

extern "C" void kernel_launch(void* const* d_in, const int* in_sizes, int n_in,
                              void* d_out, int out_size, void* d_ws, size_t ws_size,
                              hipStream_t stream) {
    const float* feats = (const float*)d_in[0];     // [B, T, 48] f32
    const float* trans = (const float*)d_in[1];     // [48, 48]   f32
    float*       out   = (float*)d_out;             // [B]        f32

    const int B = out_size;                         // 2048
    const int T = in_sizes[0] / (B * K_TAG);        // 256
    const int grid = B / 16;                        // 16 batches per wave

    if (T == 256)
        crf_fwd_wmma_kernel<256><<<grid, 32, 0, stream>>>(feats, trans, out, T);
    else
        crf_fwd_wmma_kernel<0><<<grid, 32, 0, stream>>>(feats, trans, out, T);
}
